// RNN_6725918786207
// MI455X (gfx1250) — matrix operand order repaired and checked by reading, not compiled, not run on hardware
//
#include <hip/hip_runtime.h>
#include <hip/hip_bf16.h>

// ---------------------------------------------------------------------------
// 2-layer tanh RNN, persistent-kernel formulation for gfx1250 (MI455X).
//   B=64, T=512, D_IN=64, D_MODEL=512, D_OUT=64
// Matmuls on V_WMMA_F32_16X16X32_BF16; recurrent state in LDS; weights bf16
// [N][K] in L2 streamed via global_load_b128; bulk LDS staging via
// global_load_async_to_lds_b128 (ASYNCcnt); 4 workgroups sync per timestep
// via an agent-scope atomic barrier.
// ---------------------------------------------------------------------------

typedef __attribute__((ext_vector_type(16))) __bf16 v16bf;
typedef __attribute__((ext_vector_type(8)))  float  v8f;

#define B_    64
#define T_    512
#define DIN   64
#define DM    512
#define DOUT  64
#define NWG   4
#define TPB   512           // 16 waves of 32
#define NWAVE (TPB / 32)

union Frag { v16bf v; uint4 q[2]; };

// Opaque zero (SGPR): added to weight base pointers inside the t-loop. Keeps
// the address "based on" the global kernel arg (so loads stay global_load_*
// with SADDR, LOADcnt-only), but defeats LICM so B-tiles are re-fetched from
// L2 every step instead of being hoisted and spilled to scratch.
__device__ inline size_t opaque_zero() {
  size_t x = 0;
  asm volatile("" : "+s"(x));
  return x;
}

// CDNA5 async copy: 16B global -> LDS per lane, tracked on ASYNCcnt.
__device__ inline void async_b128(unsigned lds_off, const void* gaddr) {
  asm volatile("global_load_async_to_lds_b128 %0, %1, off"
               :: "v"(lds_off), "v"(gaddr)
               : "memory");
}
__device__ inline void wait_async0() {
  asm volatile("s_wait_asynccnt 0" ::: "memory");
}

// A-matrix fragment (16x32 bf16) from LDS, row-major with row stride `stride`.
// ISA layout: lane L -> row M = L&15; K = (L>>4)*8 + {0..7} in v[0..3],
//                                   K = (L>>4)*8 + 16 + {0..7} in v[4..7].
__device__ inline v16bf load_a_lds(const __bf16* base, int stride, int mtile,
                                   int kc, int lane) {
  const int m  = mtile * 16 + (lane & 15);
  const int kb = kc * 32 + (lane >> 4) * 8;
  const __bf16* p = base + m * stride + kb;
  Frag f;
  f.q[0] = *(const uint4*)(p);        // K = kb .. kb+7
  f.q[1] = *(const uint4*)(p + 16);   // K = kb+16 .. kb+23
  return f.v;
}

// B-matrix fragment (32x16 bf16) from global weight stored transposed [N][K].
// ISA layout: lane L -> N = L&15; K = (L>>4)*8 + {0..7}, then (L>>4)*8 + 16 + {0..7}.
__device__ inline v16bf load_b_glb(const __bf16* wt, int K, int ntile,
                                   int kc, int lane) {
  const int n  = ntile * 16 + (lane & 15);
  const int kb = kc * 32 + (lane >> 4) * 8;
  const __bf16* p = wt + n * K + kb;   // two runs of 8 bf16, 16B aligned
  Frag f;
  f.q[0] = *(const uint4*)(p);
  f.q[1] = *(const uint4*)(p + 16);
  return f.v;
}

__device__ inline v8f bwmma(v16bf a, v16bf b, v8f c) {
  return __builtin_amdgcn_wmma_f32_16x16x32_bf16(
      /*neg_a=*/false, a, /*neg_b=*/false, b,
      /*c_mod=*/(short)0, c, /*reuse_a=*/false, /*reuse_b=*/false);
}

// Device-wide barrier: monotonic counter (zeroed by prep kernel each launch).
__device__ inline void grid_barrier(unsigned* cnt, unsigned target) {
  __threadfence();            // release our h-slice stores to L2 (agent scope)
  __syncthreads();
  if (threadIdx.x == 0) {
    __hip_atomic_fetch_add(cnt, 1u, __ATOMIC_ACQ_REL, __HIP_MEMORY_SCOPE_AGENT);
    while (__hip_atomic_load(cnt, __ATOMIC_ACQUIRE,
                             __HIP_MEMORY_SCOPE_AGENT) < target) {
      __builtin_amdgcn_s_sleep(16);
    }
  }
  __syncthreads();
  __threadfence();            // acquire side for all threads
}

// ---------------------------------------------------------------------------
// Prologue: bf16-convert + transpose weights to [N][K], x to [t][b][d],
// zero h buffers and the barrier counter.
// ---------------------------------------------------------------------------
__global__ void rnn_prep(const float* __restrict__ x,
                         const float* __restrict__ Wih0,
                         const float* __restrict__ Whh0,
                         const float* __restrict__ Wih1,
                         const float* __restrict__ Whh1,
                         const float* __restrict__ Wout,
                         __bf16* __restrict__ Wih0t, __bf16* __restrict__ Whh0t,
                         __bf16* __restrict__ Wih1t, __bf16* __restrict__ Whh1t,
                         __bf16* __restrict__ Woutt, __bf16* __restrict__ xbf,
                         __bf16* __restrict__ gh0,   __bf16* __restrict__ gh1,
                         unsigned* __restrict__ cnt) {
  const size_t g      = (size_t)blockIdx.x * blockDim.x + threadIdx.x;
  const size_t stride = (size_t)gridDim.x * blockDim.x;
  if (g == 0) *cnt = 0u;

  for (size_t i = g; i < (size_t)DM * DIN; i += stride) {       // Wih0t[n][k]
    size_t n = i / DIN, k = i % DIN;
    Wih0t[i] = (__bf16)Wih0[k * DM + n];
  }
  for (size_t i = g; i < (size_t)DM * DM; i += stride) {        // 512x512 x3
    size_t n = i / DM, k = i % DM;
    Whh0t[i] = (__bf16)Whh0[k * DM + n];
    Wih1t[i] = (__bf16)Wih1[k * DM + n];
    Whh1t[i] = (__bf16)Whh1[k * DM + n];
  }
  for (size_t i = g; i < (size_t)DOUT * DM; i += stride) {      // Woutt[n][k]
    size_t n = i / DM, k = i % DM;
    Woutt[i] = (__bf16)Wout[k * DOUT + n];
  }
  for (size_t i = g; i < (size_t)T_ * B_ * DIN; i += stride) {  // x -> [t][b][d]
    size_t t = i / (B_ * DIN), r = i % (B_ * DIN), b = r / DIN, d = r % DIN;
    xbf[i] = (__bf16)x[b * ((size_t)T_ * DIN) + t * DIN + d];
  }
  for (size_t i = g; i < (size_t)B_ * DM; i += stride) {        // h state = 0
    gh0[i] = (__bf16)0.f;
    gh1[i] = (__bf16)0.f;
  }
}

// ---------------------------------------------------------------------------
// Persistent RNN kernel: 4 workgroups x 512 threads (16 wave32 each).
// ---------------------------------------------------------------------------
__global__ __launch_bounds__(TPB, 1) void rnn_persist(
    const __bf16* __restrict__ xbf,
    const __bf16* __restrict__ Wih0t, const __bf16* __restrict__ Whh0t,
    const __bf16* __restrict__ Wih1t, const __bf16* __restrict__ Whh1t,
    const __bf16* __restrict__ Woutt,
    const float* __restrict__ bih0, const float* __restrict__ bhh0,
    const float* __restrict__ bih1, const float* __restrict__ bhh1,
    const float* __restrict__ bout,
    __bf16* __restrict__ gh0, __bf16* __restrict__ gh1,
    float* __restrict__ out, unsigned* __restrict__ cnt) {
  __shared__ alignas(16) __bf16 sx [B_ * DIN];   //  8 KB  x_t
  __shared__ alignas(16) __bf16 sh0[B_ * DM];    // 64 KB  h0 (prev/new)
  __shared__ alignas(16) __bf16 sh1[B_ * DM];    // 64 KB  h1 (prev/new)

  const int tid   = threadIdx.x;
  const int lane  = tid & 31;
  const int wv    = tid >> 5;
  const int gw    = blockIdx.x * NWAVE + wv;     // 0..63 global wave id
  const int ntile = gw & 31;                     // N column tile (0..31)
  const int mpair = gw >> 5;                     // 0/1 -> M tiles {0,1}/{2,3}
  const int mt0 = mpair * 2, mt1 = mpair * 2 + 1;
  const int ln = lane & 15, lh = lane >> 4;
  const int ng = ntile * 16 + ln;                // global column for C/bias

  const unsigned sx_off  = (unsigned)(uintptr_t)(void*)sx;
  const unsigned sh0_off = (unsigned)(uintptr_t)(void*)sh0;
  const unsigned sh1_off = (unsigned)(uintptr_t)(void*)sh1;

  const float bias0 = bih0[ng] + bhh0[ng];
  const float bias1 = bih1[ng] + bhh1[ng];
  // output tiles: 4x4 = 16 tiles spread over waves with (gw&3)==0
  const int otile = gw >> 2;
  const int omt = otile >> 2, ont = otile & 3;
  const bool do_out = ((gw & 3) == 0) && (otile < 16);
  const float biaso = bout[ont * 16 + ln];

  // initial LDS h state (zeros from prep): 4096 x 16B per buffer, async DMA
  for (int j = 0; j < 8; ++j) {
    const int idx = tid + j * TPB;
    async_b128(sh0_off + idx * 16, (const char*)gh0 + idx * 16);
    async_b128(sh1_off + idx * 16, (const char*)gh1 + idx * 16);
  }
  wait_async0();
  __syncthreads();

  unsigned bar = 0;
  for (int t = 0; t < T_; ++t) {
    // Opaque zero offset: addresses stay rooted at the global args (SADDR
    // global_load_b128) but are not loop-invariant -> no hoist, no spills.
    const size_t zo = opaque_zero();
    const __bf16* wih0 = Wih0t + zo;
    const __bf16* whh0 = Whh0t + zo;
    const __bf16* wih1 = Wih1t + zo;
    const __bf16* whh1 = Whh1t + zo;
    const __bf16* wout = Woutt + zo;

    // stage x_t (8 KB): one async 16B per thread
    async_b128(sx_off + tid * 16,
               (const char*)(xbf + (size_t)t * B_ * DIN) + tid * 16);
    wait_async0();
    __syncthreads();

    // ---------------- layer 0: h0 = tanh(x@Wih0 + h0@Whh0 + b) -------------
    v8f acc0, acc1;
    for (int r = 0; r < 8; ++r) { acc0[r] = bias0; acc1[r] = bias0; }
    for (int kc = 0; kc < DIN / 32; ++kc) {
      v16bf b = load_b_glb(wih0, DIN, ntile, kc, lane);
      acc0 = bwmma(load_a_lds(sx, DIN, mt0, kc, lane), b, acc0);
      acc1 = bwmma(load_a_lds(sx, DIN, mt1, kc, lane), b, acc1);
    }
    for (int kc = 0; kc < DM / 32; ++kc) {
      v16bf b = load_b_glb(whh0, DM, ntile, kc, lane);
      acc0 = bwmma(load_a_lds(sh0, DM, mt0, kc, lane), b, acc0);
      acc1 = bwmma(load_a_lds(sh0, DM, mt1, kc, lane), b, acc1);
    }
    for (int r = 0; r < 8; ++r) {
      gh0[(mt0 * 16 + lh * 8 + r) * DM + ng] = (__bf16)tanhf(acc0[r]);
      gh0[(mt1 * 16 + lh * 8 + r) * DM + ng] = (__bf16)tanhf(acc1[r]);
    }
    bar += NWG;
    grid_barrier(cnt, bar);
    // reload full h0 into LDS (async DMA, own-wave wait + WG barrier)
    for (int j = 0; j < 8; ++j) {
      const int idx = tid + j * TPB;
      async_b128(sh0_off + idx * 16, (const char*)gh0 + idx * 16);
    }
    wait_async0();
    __syncthreads();

    // ---------------- layer 1: h1 = tanh(h0@Wih1 + h1@Whh1 + b) ------------
    for (int r = 0; r < 8; ++r) { acc0[r] = bias1; acc1[r] = bias1; }
    for (int kc = 0; kc < DM / 32; ++kc) {
      v16bf b = load_b_glb(wih1, DM, ntile, kc, lane);
      acc0 = bwmma(load_a_lds(sh0, DM, mt0, kc, lane), b, acc0);
      acc1 = bwmma(load_a_lds(sh0, DM, mt1, kc, lane), b, acc1);
    }
    for (int kc = 0; kc < DM / 32; ++kc) {
      v16bf b = load_b_glb(whh1, DM, ntile, kc, lane);
      acc0 = bwmma(load_a_lds(sh1, DM, mt0, kc, lane), b, acc0);
      acc1 = bwmma(load_a_lds(sh1, DM, mt1, kc, lane), b, acc1);
    }
    for (int r = 0; r < 8; ++r) {
      gh1[(mt0 * 16 + lh * 8 + r) * DM + ng] = (__bf16)tanhf(acc0[r]);
      gh1[(mt1 * 16 + lh * 8 + r) * DM + ng] = (__bf16)tanhf(acc1[r]);
    }
    bar += NWG;
    grid_barrier(cnt, bar);
    // reload full h1 into LDS
    for (int j = 0; j < 8; ++j) {
      const int idx = tid + j * TPB;
      async_b128(sh1_off + idx * 16, (const char*)gh1 + idx * 16);
    }
    wait_async0();
    __syncthreads();

    // ---------------- output: out_t = h1 @ Wout + bout ---------------------
    if (do_out) {  // wave-uniform branch: EXEC stays all-ones for WMMA
      v8f acco;
      for (int r = 0; r < 8; ++r) acco[r] = biaso;
      for (int kc = 0; kc < DM / 32; ++kc) {
        v16bf b = load_b_glb(wout, DM, ont, kc, lane);
        acco = bwmma(load_a_lds(sh1, DM, omt, kc, lane), b, acco);
      }
      for (int r = 0; r < 8; ++r) {
        const int m = omt * 16 + lh * 8 + r;   // batch row
        const int n = ont * 16 + ln;           // output feature
        out[(size_t)m * (T_ * DOUT) + (size_t)t * DOUT + n] = acco[r];
      }
    }
  }
}

// ---------------------------------------------------------------------------
extern "C" void kernel_launch(void* const* d_in, const int* in_sizes, int n_in,
                              void* d_out, int out_size, void* d_ws,
                              size_t ws_size, hipStream_t stream) {
  const float* x    = (const float*)d_in[0];
  const float* Wih0 = (const float*)d_in[1];
  const float* bih0 = (const float*)d_in[2];
  const float* Whh0 = (const float*)d_in[3];
  const float* bhh0 = (const float*)d_in[4];
  const float* Wih1 = (const float*)d_in[5];
  const float* bih1 = (const float*)d_in[6];
  const float* Whh1 = (const float*)d_in[7];
  const float* bhh1 = (const float*)d_in[8];
  const float* Wout = (const float*)d_in[9];
  const float* bout = (const float*)d_in[10];

  char* ws = (char*)d_ws;
  size_t o = 0;
  auto take = [&](size_t bytes) {
    size_t r = o;
    o += (bytes + 255) & ~(size_t)255;
    return r;
  };
  unsigned* cnt   = (unsigned*)(ws + take(256));
  __bf16* Wih0t   = (__bf16*)(ws + take((size_t)DM * DIN * 2));
  __bf16* Whh0t   = (__bf16*)(ws + take((size_t)DM * DM * 2));
  __bf16* Wih1t   = (__bf16*)(ws + take((size_t)DM * DM * 2));
  __bf16* Whh1t   = (__bf16*)(ws + take((size_t)DM * DM * 2));
  __bf16* Woutt   = (__bf16*)(ws + take((size_t)DOUT * DM * 2));
  __bf16* gh0     = (__bf16*)(ws + take((size_t)B_ * DM * 2));
  __bf16* gh1     = (__bf16*)(ws + take((size_t)B_ * DM * 2));
  __bf16* xbf     = (__bf16*)(ws + take((size_t)T_ * B_ * DIN * 2));

  rnn_prep<<<1024, 256, 0, stream>>>(x, Wih0, Whh0, Wih1, Whh1, Wout, Wih0t,
                                     Whh0t, Wih1t, Whh1t, Woutt, xbf, gh0, gh1,
                                     cnt);
  rnn_persist<<<NWG, TPB, 0, stream>>>(xbf, Wih0t, Whh0t, Wih1t, Whh1t, Woutt,
                                       bih0, bhh0, bih1, bhh1, bout, gh0, gh1,
                                       (float*)d_out, cnt);
}
